// GATEncoder_68728066671117
// MI455X (gfx1250) — matrix operand order repaired and checked by reading, not compiled, not run on hardware
//
#include <hip/hip_runtime.h>
#include <hip/hip_bf16.h>
#include <math.h>

typedef __attribute__((ext_vector_type(16))) _Float16 v16h;
typedef __attribute__((ext_vector_type(8)))  _Float16 v8h;
typedef __attribute__((ext_vector_type(8)))  float    v8f;

#define NN 50000
#define NE 800000
#define NET (NE + NN)          // edges + self loops
#define NG 64

// Monotonic uint encoding for float atomicMax
__device__ __forceinline__ unsigned f32_enc(float f) {
  unsigned u = __float_as_uint(f);
  return (u & 0x80000000u) ? ~u : (u | 0x80000000u);
}
__device__ __forceinline__ float f32_dec(unsigned e) {
  return (e & 0x80000000u) ? __uint_as_float(e & 0x7FFFFFFFu) : __uint_as_float(~e);
}
#define ENC_NEG_INF 0x007FFFFFu   // f32_enc(-inf)

// ---------------- init kernels ----------------
__global__ void k_zero_f32(float* __restrict__ p, long long n) {
  long long t = (long long)blockIdx.x * blockDim.x + threadIdx.x;
  if (t < n) p[t] = 0.0f;
}
__global__ void k_fill_u32(unsigned* __restrict__ p, unsigned v, long long n) {
  long long t = (long long)blockIdx.x * blockDim.x + threadIdx.x;
  if (t < n) p[t] = v;
}

// ---------------- WMMA GEMM: C[NN,ncols] = A[NN,K] @ W[K,ncols] ----------------
// Block: 128 threads = 4 waves; block tile = 16 rows x 64 cols; wave tile 16x16.
// W sub-tile staged transposed in LDS as f16, row pitch K+8 halves (conflict-free).
template<int K>
__global__ __launch_bounds__(128)
void k_gemm_wmma(const float* __restrict__ A, const float* __restrict__ W,
                 float* __restrict__ C, int ncols) {
  __shared__ _Float16 sW[64 * (K + 8)];
  const int tid  = threadIdx.x;
  const int lane = tid & 31;
  const int w    = tid >> 5;
  const int row0 = blockIdx.x * 16;
  const int col0 = blockIdx.y * 64;

  for (int idx = tid; idx < K * 64; idx += 128) {
    int k = idx >> 6, c = idx & 63;
    sW[c * (K + 8) + k] = (_Float16)W[k * ncols + col0 + c];
  }
  __syncthreads();

  const int m      = row0 + (lane & 15);
  const int khalf  = (lane >> 4) * 8;           // A fragment K sub-offset
  const int kbB    = (lane >> 4) * 8;           // B fragment K sub-offset
  const float* arow = A + (size_t)m * K;
  const int cIdx   = (w * 16 + (lane & 15)) * (K + 8);

  v8f acc = {};
  #pragma unroll
  for (int k0 = 0; k0 < K; k0 += 32) {
    if (k0 + 32 < K) __builtin_prefetch(arow + k0 + 32, 0, 1);
    // A fragment: K(j) = k0 + (j>=8?16:0) + (lane/16)*8 + (j%8)
    float4 f0 = *(const float4*)(arow + k0 + khalf);
    float4 f1 = *(const float4*)(arow + k0 + khalf + 4);
    float4 f2 = *(const float4*)(arow + k0 + 16 + khalf);
    float4 f3 = *(const float4*)(arow + k0 + 16 + khalf + 4);
    v16h a = { (_Float16)f0.x, (_Float16)f0.y, (_Float16)f0.z, (_Float16)f0.w,
               (_Float16)f1.x, (_Float16)f1.y, (_Float16)f1.z, (_Float16)f1.w,
               (_Float16)f2.x, (_Float16)f2.y, (_Float16)f2.z, (_Float16)f2.w,
               (_Float16)f3.x, (_Float16)f3.y, (_Float16)f3.z, (_Float16)f3.w };
    // B fragment: K(j) = k0 + (j>=8?16:0) + (lane/16)*8 + (j%8) ; N = col0 + w*16 + lane%16
    v8h b0 = *(const v8h*)&sW[cIdx + k0 + kbB];
    v8h b1 = *(const v8h*)&sW[cIdx + k0 + kbB + 16];
    v16h b = __builtin_shufflevector(b0, b1, 0,1,2,3,4,5,6,7,8,9,10,11,12,13,14,15);
    acc = __builtin_amdgcn_wmma_f32_16x16x32_f16(false, a, false, b,
                                                 (short)0, acc, false, false);
  }
  // C/D: VGPR r -> M = row0 + r + (lane/16)*8 ; N = col0 + w*16 + lane%16
  const int crow = row0 + (lane >> 4) * 8;
  const int ccol = col0 + w * 16 + (lane & 15);
  #pragma unroll
  for (int r = 0; r < 8; ++r)
    C[(size_t)(crow + r) * ncols + ccol] = acc[r];
}

// ---------------- attention scalars: alpha_s/alpha_d per (node, head) ----------------
__global__ void k_alpha(const float* __restrict__ h, const float* __restrict__ a_s,
                        const float* __restrict__ a_d, float* __restrict__ as_out,
                        float* __restrict__ ad_out, int H, int Cc) {
  int t = blockIdx.x * blockDim.x + threadIdx.x;
  if (t >= NN * H) return;
  int node = t / H, hh = t - node * H;
  const float* hp  = h + (size_t)node * H * Cc + hh * Cc;
  const float* asp = a_s + hh * Cc;
  const float* adp = a_d + hh * Cc;
  float ss = 0.f, sd = 0.f;
  for (int c = 0; c < Cc; ++c) { float v = hp[c]; ss += v * asp[c]; sd += v * adp[c]; }
  as_out[t] = ss; ad_out[t] = sd;
}

__device__ __forceinline__ void edge_sd(const int* ei, int e, int& src, int& dst) {
  if (e < NE) { src = ei[e]; dst = ei[NE + e]; }
  else        { src = e - NE; dst = e - NE; }          // self loops appended
}

// ---------------- edge pass 1: segment max of leaky_relu(as[src]+ad[dst]) ----------------
__global__ void k_edge_max(const int* __restrict__ ei, const float* __restrict__ as,
                           const float* __restrict__ ad, unsigned* __restrict__ menc, int H) {
  long long t = (long long)blockIdx.x * blockDim.x + threadIdx.x;
  if (t >= (long long)NET * H) return;
  int e = (int)(t / H), hh = (int)(t - (long long)e * H);
  int src, dst; edge_sd(ei, e, src, dst);
  float v = as[(size_t)src * H + hh] + ad[(size_t)dst * H + hh];
  v = v > 0.f ? v : 0.2f * v;
  atomicMax(&menc[(size_t)dst * H + hh], f32_enc(v));
}

// ---------------- edge pass 2: ex = exp(e - m[dst]); denom += ex ----------------
__global__ void k_edge_exp(const int* __restrict__ ei, const float* __restrict__ as,
                           const float* __restrict__ ad, const unsigned* __restrict__ menc,
                           float* __restrict__ exb, float* __restrict__ den, int H) {
  long long t = (long long)blockIdx.x * blockDim.x + threadIdx.x;
  if (t >= (long long)NET * H) return;
  int e = (int)(t / H), hh = (int)(t - (long long)e * H);
  int src, dst; edge_sd(ei, e, src, dst);
  float v = as[(size_t)src * H + hh] + ad[(size_t)dst * H + hh];
  v = v > 0.f ? v : 0.2f * v;
  unsigned me = menc[(size_t)dst * H + hh];
  float m = (me == ENC_NEG_INF) ? 0.f : f32_dec(me);
  float ev = expf(v - m);
  exb[(size_t)e * H + hh] = ev;
  atomicAdd(&den[(size_t)dst * H + hh], ev);
}

// ---------------- edge pass 3: out[dst] += (ex/denom) * h[src], per channel ----------------
__global__ void k_edge_agg(const int* __restrict__ ei, const float* __restrict__ exb,
                           const float* __restrict__ den, const float* __restrict__ hsrc,
                           float* __restrict__ out, int H, int HC) {
  long long t = (long long)blockIdx.x * blockDim.x + threadIdx.x;
  if (t >= (long long)NET * HC) return;
  int e  = (int)(t / HC);
  int cc = (int)(t - (long long)e * HC);
  int src, dst; edge_sd(ei, e, src, dst);
  int hh = cc / (HC / H);
  float alpha = exb[(size_t)e * H + hh] / (den[(size_t)dst * H + hh] + 1e-16f);
  atomicAdd(&out[(size_t)dst * HC + cc], alpha * hsrc[(size_t)src * HC + cc]);
}

// ---------------- bias + optional ELU, in place ----------------
__global__ void k_bias_act(float* __restrict__ x, const float* __restrict__ b,
                           int HC, int elu) {
  long long t = (long long)blockIdx.x * blockDim.x + threadIdx.x;
  if (t >= (long long)NN * HC) return;
  float v = x[t] + b[(int)(t % HC)];
  if (elu) v = v > 0.f ? v : expm1f(v);
  x[t] = v;
}

// ---------------- pooling ----------------
__global__ void k_pool_accum(const float* __restrict__ emb, const int* __restrict__ batch,
                             float* __restrict__ sums) {
  long long t = (long long)blockIdx.x * blockDim.x + threadIdx.x;
  if (t >= (long long)NN * 128) return;
  int n = (int)(t >> 7), c = (int)(t & 127);
  atomicAdd(&sums[(size_t)batch[n] * 128 + c], emb[t]);
}
__global__ void k_pool_count(const int* __restrict__ batch, float* __restrict__ cnt) {
  int t = blockIdx.x * blockDim.x + threadIdx.x;
  if (t < NN) atomicAdd(&cnt[batch[t]], 1.0f);
}
__global__ void k_pool_mlp(const float* __restrict__ sums, const float* __restrict__ cnt,
                           const float* __restrict__ Wp, const float* __restrict__ bp,
                           float* __restrict__ out) {
  int g = blockIdx.x, c = threadIdx.x;
  float inv = 1.0f / fmaxf(cnt[g], 1.0f);
  float s = bp[c];
  for (int k = 0; k < 128; ++k) s += sums[g * 128 + k] * inv * Wp[k * 128 + c];
  out[g * 128 + c] = fmaxf(s, 0.0f);
}

static inline unsigned cdiv(long long a, long long b) { return (unsigned)((a + b - 1) / b); }

extern "C" void kernel_launch(void* const* d_in, const int* in_sizes, int n_in,
                              void* d_out, int out_size, void* d_ws, size_t ws_size,
                              hipStream_t stream) {
  const float* x   = (const float*)d_in[0];
  const int*   ei  = (const int*)d_in[1];
  const int*   bat = (const int*)d_in[2];
  const float* W1  = (const float*)d_in[3];
  const float* as1 = (const float*)d_in[4];
  const float* ad1 = (const float*)d_in[5];
  const float* b1  = (const float*)d_in[6];
  const float* W2  = (const float*)d_in[7];
  const float* as2 = (const float*)d_in[8];
  const float* ad2 = (const float*)d_in[9];
  const float* b2  = (const float*)d_in[10];
  const float* Wp  = (const float*)d_in[11];
  const float* bp  = (const float*)d_in[12];
  float* out = (float*)d_out;                  // [NN*128 node emb | NG*128 graph emb]

  float* ws = (float*)d_ws;
  size_t off = 0;
  float*    h1    = ws + off; off += (size_t)NN * 256;
  float*    hact  = ws + off; off += (size_t)NN * 256;
  float*    aS    = ws + off; off += (size_t)NN * 8;
  float*    aD    = ws + off; off += (size_t)NN * 8;
  unsigned* menc  = (unsigned*)(ws + off); off += (size_t)NN * 8;
  float*    den   = ws + off; off += (size_t)NN * 8;
  float*    exb   = ws + off; off += (size_t)NET * 8;
  float*    psum  = ws + off; off += (size_t)NG * 128;
  float*    pcnt  = ws + off; off += NG;
  // layer-2 aliases over layer-1 scratch (sizes strictly smaller)
  float*    h2    = h1;    // NN*128
  float*    aS2   = aS;    // NN
  float*    aD2   = aD;
  unsigned* menc2 = menc;
  float*    den2  = den;
  float*    ex2   = exb;

  const int T = 256;

  // ---- layer 1 init ----
  k_zero_f32<<<cdiv((long long)NN * 256, T), T, 0, stream>>>(hact, (long long)NN * 256);
  k_zero_f32<<<cdiv((long long)NN * 8,   T), T, 0, stream>>>(den,  (long long)NN * 8);
  k_fill_u32<<<cdiv((long long)NN * 8,   T), T, 0, stream>>>(menc, ENC_NEG_INF, (long long)NN * 8);

  // ---- layer 1: h1 = x @ W1 (WMMA) ----
  k_gemm_wmma<128><<<dim3(NN / 16, 256 / 64), 128, 0, stream>>>(x, W1, h1, 256);
  k_alpha<<<cdiv((long long)NN * 8, T), T, 0, stream>>>(h1, as1, ad1, aS, aD, 8, 32);
  k_edge_max<<<cdiv((long long)NET * 8, T), T, 0, stream>>>(ei, aS, aD, menc, 8);
  k_edge_exp<<<cdiv((long long)NET * 8, T), T, 0, stream>>>(ei, aS, aD, menc, exb, den, 8);
  k_edge_agg<<<cdiv((long long)NET * 256, T), T, 0, stream>>>(ei, exb, den, h1, hact, 8, 256);
  k_bias_act<<<cdiv((long long)NN * 256, T), T, 0, stream>>>(hact, b1, 256, 1);

  // ---- layer 2 init ----
  k_zero_f32<<<cdiv((long long)NN * 128, T), T, 0, stream>>>(out, (long long)NN * 128);
  k_zero_f32<<<cdiv((long long)NN, T), T, 0, stream>>>(den2, (long long)NN);
  k_fill_u32<<<cdiv((long long)NN, T), T, 0, stream>>>(menc2, ENC_NEG_INF, (long long)NN);

  // ---- layer 2: h2 = hact @ W2 (WMMA) ----
  k_gemm_wmma<256><<<dim3(NN / 16, 128 / 64), 128, 0, stream>>>(hact, W2, h2, 128);
  k_alpha<<<cdiv((long long)NN, T), T, 0, stream>>>(h2, as2, ad2, aS2, aD2, 1, 128);
  k_edge_max<<<cdiv((long long)NET, T), T, 0, stream>>>(ei, aS2, aD2, menc2, 1);
  k_edge_exp<<<cdiv((long long)NET, T), T, 0, stream>>>(ei, aS2, aD2, menc2, ex2, den2, 1);
  k_edge_agg<<<cdiv((long long)NET * 128, T), T, 0, stream>>>(ei, ex2, den2, h2, out, 1, 128);
  k_bias_act<<<cdiv((long long)NN * 128, T), T, 0, stream>>>(out, b2, 128, 0);

  // ---- pooling + MLP ----
  k_zero_f32<<<cdiv((long long)NG * 129, T), T, 0, stream>>>(psum, (long long)NG * 129);
  k_pool_accum<<<cdiv((long long)NN * 128, T), T, 0, stream>>>(out, bat, psum);
  k_pool_count<<<cdiv((long long)NN, T), T, 0, stream>>>(bat, pcnt);
  k_pool_mlp<<<NG, 128, 0, stream>>>(psum, pcnt, Wp, bp, out + (size_t)NN * 128);
}
